// Molmo2Attention_49589692399824
// MI455X (gfx1250) — matrix operand order repaired and checked by reading, not compiled, not run on hardware
//
#include <hip/hip_runtime.h>

#define Hh   16
#define KVHh 4
#define Dh   128
#define HIDh 2048
#define QKVN (Hh*Dh + 2*KVHh*Dh)   // 3072

typedef __attribute__((ext_vector_type(16))) __bf16 bf16x16;
typedef __attribute__((ext_vector_type(8)))  float  floatx8;

union Frag {
    bf16x16 v;
    uint4   q[2];
    __bf16  h[16];
};
union Pack8 {
    uint4  q;
    __bf16 h[8];
};

__device__ __forceinline__ __bf16 f2bf(float f) {
    unsigned int u = __builtin_bit_cast(unsigned int, f);
    unsigned int r = (u + 0x7FFFu + ((u >> 16) & 1u)) >> 16;
    unsigned short s = (unsigned short)r;
    return __builtin_bit_cast(__bf16, s);
}

// ---------------------------------------------------------------------------
// f32 -> bf16 conversion
// ---------------------------------------------------------------------------
__global__ __launch_bounds__(256) void cvt_bf16(const float* __restrict__ in,
                                                __bf16* __restrict__ out, long n) {
    long i = (long)blockIdx.x * blockDim.x + threadIdx.x;
    if (i < n) out[i] = f2bf(in[i]);
}

// ---------------------------------------------------------------------------
// Tiled bf16 GEMM: C(f32, MxN) = A(bf16, MxK, row-major) x B(bf16, KxN, row-major)
// BM=BN=128, BK=32, 256 threads = 8 waves (2 along M x 4 along N),
// each wave computes 64x32 = 4x2 tiles of 16x16 via v_wmma_f32_16x16x32_bf16.
// M, N, K must be multiples of 128/128/32 (true for all uses here).
// ---------------------------------------------------------------------------
__global__ __launch_bounds__(256) void gemm_bf16f32(const __bf16* __restrict__ A,
                                                    const __bf16* __restrict__ B,
                                                    float* __restrict__ C,
                                                    int M, int N, int K) {
    __shared__ __align__(16) __bf16 As[128 * 40];   // [row][k], padded stride 40
    __shared__ __align__(16) __bf16 Bt[128 * 40];   // [n][k], transposed, stride 40

    const int tid  = threadIdx.x;
    const int lane = tid & 31;
    const int lo   = lane & 15;
    const int hi   = lane >> 4;
    const int wave = tid >> 5;
    const int wm   = wave & 1;        // 2 waves along M (64 rows each)
    const int wn   = wave >> 1;       // 4 waves along N (32 cols each)
    const long bM  = (long)blockIdx.y * 128;
    const long bN  = (long)blockIdx.x * 128;

    floatx8 acc[4][2];
    const floatx8 zero = {0, 0, 0, 0, 0, 0, 0, 0};
#pragma unroll
    for (int i = 0; i < 4; ++i)
#pragma unroll
        for (int j = 0; j < 2; ++j) acc[i][j] = zero;

    for (int k0 = 0; k0 < K; k0 += 32) {
        // --- stage A tile 128x32 (row-major) ---
#pragma unroll
        for (int i = 0; i < 2; ++i) {
            int c   = tid + i * 256;           // 512 chunks of 8 bf16
            int row = c >> 2;
            int cc  = (c & 3) * 8;
            const __bf16* src = A + (bM + row) * (long)K + k0 + cc;
            uint4 v = *reinterpret_cast<const uint4*>(src);
            __builtin_prefetch(src + 32, 0, 1);   // next BK slab -> global_prefetch
            *reinterpret_cast<uint4*>(&As[row * 40 + cc]) = v;
        }
        // --- stage B tile 32x128, transposed into Bt[n][k] ---
#pragma unroll
        for (int i = 0; i < 2; ++i) {
            int c  = tid + i * 256;            // 512 chunks of 8 bf16
            int kr = c >> 4;
            int nc = (c & 15) * 8;
            Pack8 p;
            p.q = *reinterpret_cast<const uint4*>(B + (long)(k0 + kr) * N + bN + nc);
#pragma unroll
            for (int t = 0; t < 8; ++t) Bt[(nc + t) * 40 + kr] = p.h[t];
        }
        __syncthreads();

        // --- fragments ---
        Frag aF[4], bF[2];
#pragma unroll
        for (int mt = 0; mt < 4; ++mt) {
            int r    = wm * 64 + mt * 16 + lo;     // A: lane holds row M = lane%16
            int base = r * 40 + hi * 8;            // k = hi*8 + e(0..7), +16 for e>=8
            aF[mt].q[0] = *reinterpret_cast<const uint4*>(&As[base]);
            aF[mt].q[1] = *reinterpret_cast<const uint4*>(&As[base + 16]);
        }
#pragma unroll
        for (int nt = 0; nt < 2; ++nt) {
            int n    = wn * 32 + nt * 16 + lo;     // B: lane holds col N = lane%16
            int base = n * 40 + hi * 8;            // k = hi*8 + e(0..7), +16 for e>=8
            bF[nt].q[0] = *reinterpret_cast<const uint4*>(&Bt[base]);
            bF[nt].q[1] = *reinterpret_cast<const uint4*>(&Bt[base + 16]);
        }

#pragma unroll
        for (int mt = 0; mt < 4; ++mt)
#pragma unroll
            for (int nt = 0; nt < 2; ++nt)
                acc[mt][nt] = __builtin_amdgcn_wmma_f32_16x16x32_bf16(
                    false, aF[mt].v, false, bF[nt].v, (short)0, acc[mt][nt], false, false);
        __syncthreads();
    }

    // --- epilogue: C layout row = r + 8*hi, col = lo ---
#pragma unroll
    for (int mt = 0; mt < 4; ++mt)
#pragma unroll
        for (int nt = 0; nt < 2; ++nt)
#pragma unroll
            for (int r = 0; r < 8; ++r) {
                long row = bM + wm * 64 + mt * 16 + r + 8 * hi;
                long col = bN + wn * 32 + nt * 16 + lo;
                C[row * (long)N + col] = acc[mt][nt][r];
            }
}

// ---------------------------------------------------------------------------
// Fused RMSNorm + RoPE + bf16 pack.
// Grid: (T, 3), 256 threads = 8 waves; unit = by*8 + wave in [0,24):
//   unit 0..15 -> q head, 16..19 -> k head, 20..23 -> v head (passthrough).
// One wave per (t, head-row); 4 contiguous elements per lane (d = lane*4+j).
// ---------------------------------------------------------------------------
__global__ __launch_bounds__(256) void normrope(const float* __restrict__ qkv,
                                                const float* __restrict__ qw,
                                                const float* __restrict__ kw,
                                                const int* __restrict__ pos,
                                                __bf16* __restrict__ qb,
                                                __bf16* __restrict__ kb,
                                                __bf16* __restrict__ vb) {
    const int t    = blockIdx.x;
    const int unit = blockIdx.y * 8 + (threadIdx.x >> 5);
    const int lane = threadIdx.x & 31;
    const float* row = qkv + (long)t * QKVN;

    int colbase;
    if (unit < 16)      colbase = unit * Dh;
    else if (unit < 20) colbase = Hh * Dh + (unit - 16) * Dh;
    else                colbase = Hh * Dh + KVHh * Dh + (unit - 20) * Dh;

    float x[4];
#pragma unroll
    for (int j = 0; j < 4; ++j) x[j] = row[colbase + lane * 4 + j];

    if (unit < 20) {
        // RMSNorm over D=128 (whole wave)
        float ss = x[0]*x[0] + x[1]*x[1] + x[2]*x[2] + x[3]*x[3];
#pragma unroll
        for (int d = 16; d >= 1; d >>= 1) ss += __shfl_xor(ss, d, 32);
        float inv = rsqrtf(ss * (1.0f / Dh) + 1e-6f);
        const float* w = (unit < 16) ? qw : kw;
#pragma unroll
        for (int j = 0; j < 4; ++j) x[j] = x[j] * inv * w[lane * 4 + j];

        // RoPE: pair (d, d+64) lives on lanes l and l^16
        float p = (float)pos[t];
        float y[4];
#pragma unroll
        for (int j = 0; j < 4; ++j) {
            int d = lane * 4 + j;
            int i = d & 63;
            float fr = p * powf(1.0e6f, -(float)i / 64.0f);
            float c = __cosf(fr), s = __sinf(fr);
            float other = __shfl_xor(x[j], 16, 32);
            y[j] = (d < 64) ? (x[j] * c - other * s) : (x[j] * c + other * s);
        }
#pragma unroll
        for (int j = 0; j < 4; ++j) x[j] = y[j];
    }

    __bf16* dst;
    if (unit < 16)      dst = qb + ((long)t * Hh   + unit)        * Dh;
    else if (unit < 20) dst = kb + ((long)t * KVHh + (unit - 16)) * Dh;
    else                dst = vb + ((long)t * KVHh + (unit - 20)) * Dh;
#pragma unroll
    for (int j = 0; j < 4; ++j) dst[lane * 4 + j] = f2bf(x[j]);
}

// ---------------------------------------------------------------------------
// Causal flash attention, one wave per (head, 16-query tile), 32-key blocks.
//   S(16x32) = Q(16x128) Kᵀ  : 8 x wmma_f32_16x16x32_bf16
//   online softmax per row (16-lane shfl reductions)
//   O += P(16x32) V(32x128)  : 8 x wmma (P re-laid out C->A through LDS)
// ---------------------------------------------------------------------------
__global__ __launch_bounds__(32) void flash16(const __bf16* __restrict__ Q,
                                              const __bf16* __restrict__ Kt,
                                              const __bf16* __restrict__ V,
                                              __bf16* __restrict__ Out) {
    const int h    = blockIdx.x;
    const int qt   = blockIdx.y;
    const int lane = threadIdx.x;
    const int lo   = lane & 15;
    const int hi   = lane >> 4;
    const int kvh  = h >> 2;          // H/KVH = 4
    const int qbase = qt * 16;

    __shared__ __align__(16) __bf16 Pl[16 * 32];

    // Q fragments: 4 chunks of 16x32 along D (A-matrix layout, b128 loads)
    Frag qF[4];
#pragma unroll
    for (int kk = 0; kk < 4; ++kk) {
        const __bf16* qp = Q + ((long)(qbase + lo) * Hh + h) * Dh + kk * 32 + hi * 8;
        qF[kk].q[0] = *reinterpret_cast<const uint4*>(qp);
        qF[kk].q[1] = *reinterpret_cast<const uint4*>(qp + 16);
    }

    const floatx8 zero = {0, 0, 0, 0, 0, 0, 0, 0};
    floatx8 O[8];
    float m[8], l[8];
#pragma unroll
    for (int dt = 0; dt < 8; ++dt) O[dt] = zero;
#pragma unroll
    for (int r = 0; r < 8; ++r) { m[r] = -1.0e30f; l[r] = 0.0f; }

    const float scale = 0.08838834764831845f;   // 1/sqrt(128)
    const int nkb = (qbase + 16 + 31) >> 5;     // key blocks covering keys <= qbase+15

    for (int b = 0; b < nkb; ++b) {
        const int kb0 = b * 32;

        // ---- S = Q Kᵀ (two 16x16 accumulators: keys kb0+lo, kb0+16+lo) ----
        floatx8 S0 = zero, S1 = zero;
#pragma unroll
        for (int kk = 0; kk < 4; ++kk) {
            Frag b0, b1;   // B layout: lane = key col, d = hi*8 + e(0..7), +16 for e>=8 (K-dim)
            const __bf16* kp0 = Kt + ((long)(kb0 + lo)      * KVHh + kvh) * Dh + kk * 32 + hi * 8;
            const __bf16* kp1 = Kt + ((long)(kb0 + 16 + lo) * KVHh + kvh) * Dh + kk * 32 + hi * 8;
            b0.q[0] = *reinterpret_cast<const uint4*>(kp0);
            b0.q[1] = *reinterpret_cast<const uint4*>(kp0 + 16);
            b1.q[0] = *reinterpret_cast<const uint4*>(kp1);
            b1.q[1] = *reinterpret_cast<const uint4*>(kp1 + 16);
            S0 = __builtin_amdgcn_wmma_f32_16x16x32_bf16(false, qF[kk].v, false, b0.v,
                                                         (short)0, S0, false, false);
            S1 = __builtin_amdgcn_wmma_f32_16x16x32_bf16(false, qF[kk].v, false, b1.v,
                                                         (short)0, S1, false, false);
        }

        // ---- scale + causal mask + online softmax ----
#pragma unroll
        for (int r = 0; r < 8; ++r) {
            int q_g = qbase + r + 8 * hi;
            float s0 = S0[r] * scale;
            float s1 = S1[r] * scale;
            if (kb0 + lo > q_g)      s0 = -3.0e38f;
            if (kb0 + 16 + lo > q_g) s1 = -3.0e38f;

            float mx = fmaxf(s0, s1);
#pragma unroll
            for (int d = 8; d >= 1; d >>= 1) mx = fmaxf(mx, __shfl_xor(mx, d, 16));
            float m_new = fmaxf(m[r], mx);

            float p0 = __expf(s0 - m_new);
            float p1 = __expf(s1 - m_new);
            float rs = p0 + p1;
#pragma unroll
            for (int d = 8; d >= 1; d >>= 1) rs += __shfl_xor(rs, d, 16);

            float co = __expf(m[r] - m_new);
            l[r] = l[r] * co + rs;
            m[r] = m_new;
#pragma unroll
            for (int dt = 0; dt < 8; ++dt) O[dt][r] *= co;

            int prow = r + 8 * hi;
            Pl[prow * 32 + lo]      = f2bf(p0);
            Pl[prow * 32 + 16 + lo] = f2bf(p1);
        }
        __syncthreads();

        // ---- P re-laid out as A fragment (16x32) ----
        Frag pF;
        {
            int base = lo * 32 + hi * 8;
            pF.q[0] = *reinterpret_cast<const uint4*>(&Pl[base]);
            pF.q[1] = *reinterpret_cast<const uint4*>(&Pl[base + 16]);
        }

        // ---- O += P V  (8 d-column tiles of 16) ----
#pragma unroll
        for (int dt = 0; dt < 8; ++dt) {
            Frag vF;   // B layout: lane = d col, key = hi*8 + e(0..7), +16 for e>=8 (K-dim)
#pragma unroll
            for (int e = 0; e < 16; ++e) {
                int key = kb0 + hi * 8 + e + (e & 8);
                vF.h[e] = V[((long)key * KVHh + kvh) * Dh + dt * 16 + lo];
            }
            O[dt] = __builtin_amdgcn_wmma_f32_16x16x32_bf16(false, pF.v, false, vF.v,
                                                            (short)0, O[dt], false, false);
        }
        __syncthreads();
    }

    // ---- normalize and store bf16 attn output [T][H*D] ----
#pragma unroll
    for (int dt = 0; dt < 8; ++dt)
#pragma unroll
        for (int r = 0; r < 8; ++r) {
            int q_g = qbase + r + 8 * hi;
            float v = O[dt][r] / l[r];
            Out[(long)q_g * (Hh * Dh) + h * Dh + dt * 16 + lo] = f2bf(v);
        }
}

// ---------------------------------------------------------------------------
// Host-side orchestration
// ---------------------------------------------------------------------------
extern "C" void kernel_launch(void* const* d_in, const int* in_sizes, int n_in,
                              void* d_out, int out_size, void* d_ws, size_t ws_size,
                              hipStream_t stream) {
    const float* hidden = (const float*)d_in[0];
    const float* w_qkv  = (const float*)d_in[1];
    const float* q_nw   = (const float*)d_in[2];
    const float* k_nw   = (const float*)d_in[3];
    const float* w_o    = (const float*)d_in[4];
    const int*   posits = (const int*)d_in[5];
    float* out = (float*)d_out;

    const long T = in_sizes[0] / HIDh;   // 4096

    char* ws = (char*)d_ws;
    size_t off = 0;
    auto alloc = [&](size_t bytes) -> void* {
        void* p = ws + off;
        off += (bytes + 255) & ~(size_t)255;
        return p;
    };
    __bf16* hsb  = (__bf16*)alloc((size_t)T * HIDh * 2);
    __bf16* wqb  = (__bf16*)alloc((size_t)HIDh * QKVN * 2);
    __bf16* wob  = (__bf16*)alloc((size_t)HIDh * HIDh * 2);
    float*  qkvf = (float*) alloc((size_t)T * QKVN * 4);
    __bf16* qb   = (__bf16*)alloc((size_t)T * Hh * Dh * 2);
    __bf16* kb   = (__bf16*)alloc((size_t)T * KVHh * Dh * 2);
    __bf16* vb   = (__bf16*)alloc((size_t)T * KVHh * Dh * 2);
    __bf16* ab   = (__bf16*)alloc((size_t)T * Hh * Dh * 2);

    { long n = T * HIDh;              cvt_bf16<<<dim3((n + 255) / 256), 256, 0, stream>>>(hidden, hsb, n); }
    { long n = (long)HIDh * QKVN;     cvt_bf16<<<dim3((n + 255) / 256), 256, 0, stream>>>(w_qkv,  wqb, n); }
    { long n = (long)HIDh * HIDh;     cvt_bf16<<<dim3((n + 255) / 256), 256, 0, stream>>>(w_o,    wob, n); }

    // qkv = hidden @ w_qkv   (4096 x 2048 x 3072)
    gemm_bf16f32<<<dim3(QKVN / 128, T / 128), 256, 0, stream>>>(hsb, wqb, qkvf,
                                                                (int)T, QKVN, HIDh);
    // rmsnorm + rope -> bf16 q/k/v
    normrope<<<dim3((unsigned)T, 3), 256, 0, stream>>>(qkvf, q_nw, k_nw, posits, qb, kb, vb);

    // causal flash attention
    flash16<<<dim3(Hh, (unsigned)(T / 16)), 32, 0, stream>>>(qb, kb, vb, ab);

    // out = attn @ w_o   (4096 x 2048 x 2048)
    gemm_bf16f32<<<dim3(HIDh / 128, T / 128), 256, 0, stream>>>(ab, wob, out,
                                                                (int)T, HIDh, HIDh);
}
